// Decoder_16449724744798
// MI455X (gfx1250) — matrix operand order repaired and checked by reading, not compiled, not run on hardware
//
#include <hip/hip_runtime.h>
#include <math.h>

// ---------------------------------------------------------------------------
// MI455X (gfx1250) decoder step: attention + GRU + vocab softmax.
// Large GEMMs use v_wmma_f32_16x16x32_bf16 (bf16 tiles staged in LDS, f32
// accumulate). Bandwidth-bound overall (~450MB HBM @ 23.3TB/s ~ 19us); bf16
// WMMA (K=32/instr) keeps the ~55 GFLOP of matrix work far below that.
// Raggedness (K=1324) is removed by zero-padding xi and gru_k to K=1344 in
// workspace, so the GEMM hot loop is completely guard-free.
// ---------------------------------------------------------------------------

typedef __attribute__((ext_vector_type(16))) __bf16 v16bf;
typedef __attribute__((ext_vector_type(8)))  __bf16 v8bf;
typedef __attribute__((ext_vector_type(4)))  __bf16 v4bf;
typedef __attribute__((ext_vector_type(8)))  float  v8f;

#define BATCH     256
#define SEQ       128
#define ENCD      1024
#define DECU      512
#define UNITS_N   1024
#define EMBD      300
#define INDIM     1324
#define INDIM_P   1344      // INDIM padded to multiple of 32
#define VOCAB_N   32000

static __device__ __forceinline__ v16bf cat16(v8bf lo, v8bf hi) {
  return __builtin_shufflevector(lo, hi, 0,1,2,3,4,5,6,7,8,9,10,11,12,13,14,15);
}
static __device__ __forceinline__ v4bf pack4(float4 v) {
  v4bf r; r[0] = (__bf16)v.x; r[1] = (__bf16)v.y; r[2] = (__bf16)v.z; r[3] = (__bf16)v.w;
  return r;
}

// ---------------------------------------------------------------------------
// Generic GEMM: C[M,N] = A[M,K] @ B[K,N] + bias[N]   (all f32 row-major)
// Block tile 32(M) x 256(N), K-step 32. 8 waves = 2(M) x 4(N); each wave a
// 16x64 tile = 4 WMMA 16x16x32 bf16 tiles.
// Requires: M % 32 == 0, N % 256 == 0, K % 32 == 0 (pre-padded by caller).
// ---------------------------------------------------------------------------
__global__ __launch_bounds__(256) void gemm_bias_kernel(
    const float* __restrict__ A, const float* __restrict__ Bm,
    const float* __restrict__ bias, float* __restrict__ C,
    int M, int N, int K)
{
  __shared__ __align__(16) __bf16 As[32][40];   // [m][k], 80B row stride
  __shared__ __align__(16) __bf16 Bs[256][40];  // [n][k] (transposed tile)

  const int tid  = threadIdx.x;
  const int lane = tid & 31;
  const int wave = tid >> 5;
  const int m0 = blockIdx.y * 32;
  const int n0 = blockIdx.x * 256;
  const int mb = (wave >> 2) * 16;   // 0 or 16
  const int nb = (wave & 3) * 64;    // 0,64,128,192

  // fragment LDS addresses (constant across k-steps)
  const int arow  = mb + (lane & 15);
  const int koff  = (lane < 16) ? 0 : 8;   // A: K {0-7,16-23} vs {8-15,24-31}
  const int kboff = (lane < 16) ? 0 : 16;  // unused; B fragment reads with koff, same K order as A
  // A staging indices: 4 consecutive k per thread
  const int sar = tid >> 3;          // 0..31 row
  const int sac = (tid & 7) * 4;     // 0,4,..,28 k

  v8f acc[4] = {};

  for (int kb = 0; kb < K; kb += 32) {
    __syncthreads();
    // ---- stage A: 32 rows x 32 k, float4 -> v4bf -> one b64 store ----
    {
      const float4 av = *(const float4*)&A[(size_t)(m0 + sar) * K + kb + sac];
      *(v4bf*)&As[sar][sac] = pack4(av);
    }
    // ---- stage B transposed: one column per thread, 32 k, b128 stores ----
    {
      const float* bp = &Bm[(size_t)kb * N + n0 + tid];
      #pragma unroll
      for (int kk = 0; kk < 32; kk += 8) {
        v8bf pk;
        #pragma unroll
        for (int j = 0; j < 8; ++j) pk[j] = (__bf16)bp[(size_t)(kk + j) * N];
        *(v8bf*)&Bs[tid][kk] = pk;
      }
    }
    __syncthreads();

    const v16bf afrag = cat16(*(const v8bf*)&As[arow][koff],
                              *(const v8bf*)&As[arow][16 + koff]);
    #pragma unroll
    for (int nt = 0; nt < 4; ++nt) {
      const int col = nb + nt * 16 + (lane & 15);
      const v16bf bfrag = cat16(*(const v8bf*)&Bs[col][koff],
                                *(const v8bf*)&Bs[col][16 + koff]);
      acc[nt] = __builtin_amdgcn_wmma_f32_16x16x32_bf16(
          false, afrag, false, bfrag, (short)0, acc[nt], false, false);
    }
  }

  // ---- epilogue: C/D layout -> global ----
  #pragma unroll
  for (int nt = 0; nt < 4; ++nt) {
    const int col = n0 + nb + nt * 16 + (lane & 15);
    const float bv = bias[col];
    #pragma unroll
    for (int v = 0; v < 8; ++v) {
      const int row = m0 + mb + v + ((lane < 16) ? 0 : 8);
      C[(size_t)row * N + col] = acc[nt][v] + bv;
    }
  }
}

// ---------------------------------------------------------------------------
// Fused attention score: score[b,s] = V . tanh(enc[b,s]@W1 + b1 + q[b])
// One workgroup per 16 flat rows (all same b), full N=512 (8 waves x 64 cols).
// Avoids materializing the [32768,512] tanh intermediate (67MB).
// ---------------------------------------------------------------------------
__global__ __launch_bounds__(256) void attn_score_kernel(
    const float* __restrict__ enc,   // [B*S, 1024]
    const float* __restrict__ W1,    // [1024, 512]
    const float* __restrict__ b1,    // [512]
    const float* __restrict__ q,     // [B, 512]
    const float* __restrict__ Vv,    // [512]
    float* __restrict__ score)       // [B*S]
{
  __shared__ __align__(16) __bf16 As[16][40];
  __shared__ __align__(16) __bf16 Bs[512][40];
  __shared__ float sc[16];

  const int tid  = threadIdx.x;
  const int lane = tid & 31;
  const int wave = tid >> 5;
  const int row0 = blockIdx.x * 16;   // flat (b*S+s) row base; 16 | 128 so one b
  const int bidx = row0 >> 7;
  const int nb = wave * 64;

  if (tid < 16) sc[tid] = 0.0f;

  const int arow  = lane & 15;
  const int koff  = (lane < 16) ? 0 : 8;
  const int kboff = (lane < 16) ? 0 : 16;
  const int sar = tid >> 3;          // for threads < 128: 0..15
  const int sac = (tid & 7) * 4;

  v8f acc[4] = {};

  for (int kb = 0; kb < ENCD; kb += 32) {
    __syncthreads();
    // ---- stage A: 16 x 32 (threads 0..127) ----
    if (tid < 128) {
      const float4 av = *(const float4*)&enc[(size_t)(row0 + sar) * ENCD + kb + sac];
      *(v4bf*)&As[sar][sac] = pack4(av);
    }
    // ---- stage B transposed: 32 k x 512 n, 2 columns per thread ----
    #pragma unroll
    for (int cc = 0; cc < 512; cc += 256) {
      const int n = cc + tid;
      const float* bp = &W1[(size_t)kb * DECU + n];
      #pragma unroll
      for (int kk = 0; kk < 32; kk += 8) {
        v8bf pk;
        #pragma unroll
        for (int j = 0; j < 8; ++j) pk[j] = (__bf16)bp[(size_t)(kk + j) * DECU];
        *(v8bf*)&Bs[n][kk] = pk;
      }
    }
    __syncthreads();

    const v16bf afrag = cat16(*(const v8bf*)&As[arow][koff],
                              *(const v8bf*)&As[arow][16 + koff]);
    #pragma unroll
    for (int nt = 0; nt < 4; ++nt) {
      const int col = nb + nt * 16 + (lane & 15);
      const v16bf bfrag = cat16(*(const v8bf*)&Bs[col][koff],
                                *(const v8bf*)&Bs[col][16 + koff]);
      acc[nt] = __builtin_amdgcn_wmma_f32_16x16x32_bf16(
          false, afrag, false, bfrag, (short)0, acc[nt], false, false);
    }
  }

  // epilogue: tanh(+b1+q), dot with V, reduce across 512 cols via LDS
  const float* qrow = q + (size_t)bidx * DECU;
  #pragma unroll
  for (int v = 0; v < 8; ++v) {
    const int r = v + ((lane < 16) ? 0 : 8);
    float p = 0.0f;
    #pragma unroll
    for (int nt = 0; nt < 4; ++nt) {
      const int col = nb + nt * 16 + (lane & 15);
      const float t = tanhf(acc[nt][v] + b1[col] + qrow[col]);
      p += t * Vv[col];
    }
    atomicAdd(&sc[r], p);
  }
  __syncthreads();
  if (tid < 16) score[row0 + tid] = sc[tid];  // bV dropped: softmax-invariant
}

// softmax over S=128 per batch row -> attn (goes straight into d_out)
__global__ __launch_bounds__(128) void softmax_s_kernel(
    const float* __restrict__ score, float* __restrict__ attn)
{
  const int b = blockIdx.x, t = threadIdx.x;
  __shared__ float red[128];
  const float x = score[b * SEQ + t];
  red[t] = x; __syncthreads();
  for (int off = 64; off > 0; off >>= 1) {
    if (t < off) red[t] = fmaxf(red[t], red[t + off]);
    __syncthreads();
  }
  const float m = red[0]; __syncthreads();
  const float e = __expf(x - m);
  red[t] = e; __syncthreads();
  for (int off = 64; off > 0; off >>= 1) {
    if (t < off) red[t] += red[t + off];
    __syncthreads();
  }
  attn[b * SEQ + t] = e / red[0];
}

// context[b,d] = sum_s attn[b,s] * enc[b,s,d]
__global__ __launch_bounds__(256) void context_kernel(
    const float* __restrict__ attn, const float* __restrict__ enc,
    float* __restrict__ ctx)
{
  const int b = blockIdx.x, t = threadIdx.x;
  __shared__ float aw[SEQ];
  if (t < SEQ) aw[t] = attn[b * SEQ + t];
  __syncthreads();
  float a0 = 0.f, a1 = 0.f, a2 = 0.f, a3 = 0.f;
  for (int s = 0; s < SEQ; ++s) {
    const float* e = enc + ((size_t)b * SEQ + s) * ENCD;
    const float a = aw[s];
    a0 += a * e[t]; a1 += a * e[t + 256]; a2 += a * e[t + 512]; a3 += a * e[t + 768];
  }
  float* c = ctx + (size_t)b * ENCD;
  c[t] = a0; c[t + 256] = a1; c[t + 512] = a2; c[t + 768] = a3;
}

// xi = concat(context, emb[x[b]], zeros)  -> [256, 1344] (K-padded)
__global__ __launch_bounds__(256) void xi_kernel(
    const float* __restrict__ ctx, const float* __restrict__ emb,
    const int* __restrict__ x, float* __restrict__ xi)
{
  const int b = blockIdx.x;
  const int tok = x[b];
  for (int i = threadIdx.x; i < INDIM_P; i += blockDim.x) {
    float v = 0.0f;
    if (i < ENCD)       v = ctx[(size_t)b * ENCD + i];
    else if (i < INDIM) v = emb[(size_t)tok * EMBD + (i - ENCD)];
    xi[(size_t)b * INDIM_P + i] = v;
  }
}

// gru_k [1324,3072] -> zero-padded copy [1344,3072]
__global__ __launch_bounds__(256) void pad_gruk_kernel(
    const float* __restrict__ gru_k, float* __restrict__ gkp)
{
  const size_t idx = (size_t)blockIdx.x * blockDim.x + threadIdx.x;
  const int r = (int)(idx / 3072);
  gkp[idx] = (r < INDIM) ? gru_k[idx] : 0.0f;
}

// GRU gates elementwise -> h (into d_out)
__global__ __launch_bounds__(256) void gru_kernel(
    const float* __restrict__ mx, const float* __restrict__ mh,
    const float* __restrict__ state, float* __restrict__ h)
{
  const int idx = blockIdx.x * blockDim.x + threadIdx.x;  // B*UNITS
  const int b = idx >> 10, u = idx & 1023;
  const float* mxr = mx + (size_t)b * 3072;
  const float* mhr = mh + (size_t)b * 3072;
  const float z = 1.0f / (1.0f + __expf(-(mxr[u] + mhr[u])));
  const float r = 1.0f / (1.0f + __expf(-(mxr[1024 + u] + mhr[1024 + u])));
  const float hh = tanhf(mxr[2048 + u] + r * mhr[2048 + u]);
  const float st = state[idx];
  h[idx] = z * st + (1.0f - z) * hh;
}

// softmax over VOCAB=32000, in place on the logits in d_out
__global__ __launch_bounds__(256) void softmax_v_kernel(float* __restrict__ probs)
{
  const int b = blockIdx.x, t = threadIdx.x;
  float* row = probs + (size_t)b * VOCAB_N;
  __shared__ float red[256];
  float m = -INFINITY;
  for (int i = t; i < VOCAB_N; i += 256) m = fmaxf(m, row[i]);
  red[t] = m; __syncthreads();
  for (int off = 128; off > 0; off >>= 1) {
    if (t < off) red[t] = fmaxf(red[t], red[t + off]);
    __syncthreads();
  }
  m = red[0]; __syncthreads();
  float s = 0.0f;
  for (int i = t; i < VOCAB_N; i += 256) s += __expf(row[i] - m);
  red[t] = s; __syncthreads();
  for (int off = 128; off > 0; off >>= 1) {
    if (t < off) red[t] += red[t + off];
    __syncthreads();
  }
  const float inv = 1.0f / red[0];
  for (int i = t; i < VOCAB_N; i += 256) row[i] = __expf(row[i] - m) * inv;
}

// ---------------------------------------------------------------------------
extern "C" void kernel_launch(void* const* d_in, const int* in_sizes, int n_in,
                              void* d_out, int out_size, void* d_ws, size_t ws_size,
                              hipStream_t stream) {
  (void)in_sizes; (void)n_in; (void)out_size; (void)ws_size;
  const int*   x      = (const int*)  d_in[0];
  const float* state  = (const float*)d_in[1];
  const float* enc    = (const float*)d_in[2];
  const float* emb    = (const float*)d_in[3];
  const float* W1     = (const float*)d_in[4];
  const float* b1     = (const float*)d_in[5];
  const float* W2     = (const float*)d_in[6];
  const float* b2     = (const float*)d_in[7];
  const float* Vv     = (const float*)d_in[8];
  /* d_in[9] = bV: softmax-invariant, unused */
  const float* gru_k  = (const float*)d_in[10];
  const float* gru_rk = (const float*)d_in[11];
  const float* gru_b  = (const float*)d_in[12];
  const float* fc_W   = (const float*)d_in[13];
  const float* fc_b   = (const float*)d_in[14];

  float* out   = (float*)d_out;
  float* probs = out;                                   // [256, 32000]
  float* h     = out + (size_t)BATCH * VOCAB_N;         // [256, 1024]
  float* attn  = h + (size_t)BATCH * UNITS_N;           // [256, 128]

  float* ws    = (float*)d_ws;                          // ~26 MB total
  float* q     = ws;                                    // 256*512
  float* score = q     + (size_t)BATCH * DECU;          // 256*128
  float* ctx   = score + (size_t)BATCH * SEQ;           // 256*1024
  float* xi    = ctx   + (size_t)BATCH * ENCD;          // 256*1344 (padded)
  float* mx    = xi    + (size_t)BATCH * INDIM_P;       // 256*3072
  float* mh    = mx    + (size_t)BATCH * 3072;          // 256*3072
  float* gkp   = mh    + (size_t)BATCH * 3072;          // 1344*3072 (padded gru_k)

  // q = state @ W2 + b2                      [256,1024]x[1024,512]
  gemm_bias_kernel<<<dim3(DECU / 256, BATCH / 32), 256, 0, stream>>>(
      state, W2, b2, q, BATCH, DECU, UNITS_N);
  // score[b,s] = V . tanh(enc@W1 + b1 + q)   fused, 34.4 GFLOP
  attn_score_kernel<<<dim3(BATCH * SEQ / 16), 256, 0, stream>>>(
      enc, W1, b1, q, Vv, score);
  softmax_s_kernel<<<BATCH, 128, 0, stream>>>(score, attn);
  context_kernel<<<BATCH, 256, 0, stream>>>(attn, enc, ctx);
  xi_kernel<<<BATCH, 256, 0, stream>>>(ctx, emb, x, xi);
  pad_gruk_kernel<<<(INDIM_P * 3072) / 256, 256, 0, stream>>>(gru_k, gkp);
  // mx = xi @ gru_k + gru_b[0]               K padded to 1344
  gemm_bias_kernel<<<dim3(3072 / 256, BATCH / 32), 256, 0, stream>>>(
      xi, gkp, gru_b, mx, BATCH, 3072, INDIM_P);
  // mh = state @ gru_rk + gru_b[1]
  gemm_bias_kernel<<<dim3(3072 / 256, BATCH / 32), 256, 0, stream>>>(
      state, gru_rk, gru_b + 3072, mh, BATCH, 3072, UNITS_N);
  gru_kernel<<<(BATCH * UNITS_N) / 256, 256, 0, stream>>>(mx, mh, state, h);
  // logits = h @ fc_W + fc_b                 16.8 GFLOP, N=32000=125*256
  gemm_bias_kernel<<<dim3(VOCAB_N / 256, BATCH / 32), 256, 0, stream>>>(
      h, fc_W, fc_b, probs, BATCH, VOCAB_N, UNITS_N);
  softmax_v_kernel<<<BATCH, 256, 0, stream>>>(probs);
}
